// GPT2Attention_90065464197659
// MI455X (gfx1250) — compile-verified, matrix operand order checked by reading, not run on hardware
//
#include <hip/hip_runtime.h>
#include <hip/hip_bf16.h>

// ---- problem constants (GPT-2 small attention, B=8, S=1024, D=768, H=12) ----
#define BB   8
#define SS   1024
#define DD   768
#define HH   12
#define HD   64
#define N3   2304   // 3*D

typedef __attribute__((ext_vector_type(16))) __bf16 v16bf;
typedef __attribute__((ext_vector_type(8)))  __bf16 v8bf;
typedef __attribute__((ext_vector_type(8)))  float  v8f;
typedef __attribute__((ext_vector_type(4)))  unsigned int u32x4;
typedef __attribute__((ext_vector_type(8)))  int          i32x8;
typedef __attribute__((ext_vector_type(4)))  int          i32x4;

// Concatenate two 8-element bf16 chunks into one 16-element WMMA fragment.
__device__ __forceinline__ v16bf pack16(v8bf lo, v8bf hi) {
    return __builtin_shufflevector(lo, hi, 0, 1, 2, 3, 4, 5, 6, 7,
                                           8, 9, 10, 11, 12, 13, 14, 15);
}

__device__ __forceinline__ void cvt4(v16bf& a, int base, float4 f) {
    a[base + 0] = (__bf16)f.x;
    a[base + 1] = (__bf16)f.y;
    a[base + 2] = (__bf16)f.z;
    a[base + 3] = (__bf16)f.w;
}

__device__ __forceinline__ float redmax16(float v) {
    #pragma unroll
    for (int m = 1; m < 16; m <<= 1) v = fmaxf(v, __shfl_xor(v, m, 32));
    return v;
}
__device__ __forceinline__ float redsum16(float v) {
    #pragma unroll
    for (int m = 1; m < 16; m <<= 1) v += __shfl_xor(v, m, 32);
    return v;
}

// ---------------------------------------------------------------------------
// TDM: DMA a 2D bf16 tile (tile_h rows x tile_w elems, row stride = stride_e)
// from global memory into LDS at byte offset lds_off.  D# packing per
// CDNA5 ISA (group0/group1; groups 2-3 zero for 2D).
// ---------------------------------------------------------------------------
#if defined(__HIP_DEVICE_COMPILE__) && __has_builtin(__builtin_amdgcn_tensor_load_to_lds)
#define HAVE_TDM 1
__device__ __forceinline__ void tdm_load_2d_bf16(
    const __bf16* gsrc, unsigned int lds_off,
    unsigned int tensor_w, unsigned int tensor_h,
    unsigned int tile_w, unsigned int tile_h, unsigned long long stride_e)
{
    unsigned long long ga = (unsigned long long)(uintptr_t)gsrc;
    u32x4 g0;
    g0[0] = 1u;                                        // count=1 (valid D#)
    g0[1] = lds_off;                                   // lds_addr (bytes)
    g0[2] = (unsigned int)(ga & 0xffffffffu);          // global_addr[31:0]
    g0[3] = (unsigned int)((ga >> 32) & 0x01ffffffu)   // global_addr[56:32]
          | (2u << 30);                                // type=2 ("image")
    i32x8 g1;
    g1[0] = (int)(1u << 16);                           // data_size=1 -> 2 bytes
    g1[1] = (int)((tensor_w & 0xffffu) << 16);         // tensor_dim0[15:0]
    g1[2] = (int)((tensor_w >> 16) | ((tensor_h & 0xffffu) << 16));
    g1[3] = (int)((tensor_h >> 16) | (tile_w << 16));  // tile_dim0
    g1[4] = (int)(tile_h);                             // tile_dim1 (tile_dim2=0)
    g1[5] = (int)(stride_e & 0xffffffffull);           // tensor_dim0_stride lo
    g1[6] = (int)((stride_e >> 32) & 0xffffull);       // stride hi, dim1_stride=0
    g1[7] = 0;
    i32x4 z4 = {0, 0, 0, 0};
#if __clang_major__ >= 23
    i32x8 z8 = {0, 0, 0, 0, 0, 0, 0, 0};
    __builtin_amdgcn_tensor_load_to_lds(g0, g1, z4, z4, z8, 0);
#else
    __builtin_amdgcn_tensor_load_to_lds(g0, g1, z4, z4, 0);
#endif
}
#else
#define HAVE_TDM 0
#endif

// ---------------------------------------------------------------------------
// Kernel 1: qkv = x @ Wqkv + bqkv.
// Block = 256 threads (8 waves) computes a 128x64 tile of the 8192x2304
// result.  B (weights) staged transposed in LDS as bf16 so each wave's four
// B fragments are aligned ds_load_b128.  A fragments are contiguous 128-bit
// global loads + cvt_pk_bf16.  Each wave: 4 WMMAs per 32-wide K step.
// Q/K written in [B,H,S,hd] bf16; V written TRANSPOSED as [B,H,hd,S] bf16.
// ---------------------------------------------------------------------------
__global__ __launch_bounds__(256) void qkv_gemm_kernel(
    const float* __restrict__ x, const float* __restrict__ Wqkv,
    const float* __restrict__ bqkv,
    __bf16* __restrict__ qh, __bf16* __restrict__ kh, __bf16* __restrict__ vT)
{
    __shared__ __align__(16) __bf16 Bs[64 * 32];     // [n][k] transposed tile

    const int lane = threadIdx.x & 31;
    const int wid  = threadIdx.x >> 5;
    const int nn   = lane & 15;
    const int half = lane >> 4;
    const int ntile = blockIdx.x;                    // 0..35 (64 cols each)
    const int mt    = blockIdx.y;                    // 0..63 (128 rows each)

    // staging map: 32 k-rows x 64 n-cols of W, 8 floats per thread
    const int kr = threadIdx.x >> 3;                 // 0..31
    const int nc = (threadIdx.x & 7) << 3;           // 0,8,...,56
    const float* wsrc = Wqkv + (size_t)ntile * 64 + nc;

    const int arowg = mt * 128 + wid * 16 + nn;
    const float* xr = x + (size_t)arowg * DD;

    v8f acc[4];
    {
        v8f z = {};
        #pragma unroll
        for (int t = 0; t < 4; ++t) acc[t] = z;
    }

    for (int kb = 0; kb < DD; kb += 32) {
        // ---- stage B tile: coalesced f32 loads -> bf16 -> transposed LDS ----
        const float* wrow = wsrc + (size_t)(kb + kr) * N3;
        float4 w0 = *(const float4*)(wrow);
        float4 w1 = *(const float4*)(wrow + 4);
        __syncthreads();                             // prior reads complete
        Bs[(nc + 0) * 32 + kr] = (__bf16)w0.x;
        Bs[(nc + 1) * 32 + kr] = (__bf16)w0.y;
        Bs[(nc + 2) * 32 + kr] = (__bf16)w0.z;
        Bs[(nc + 3) * 32 + kr] = (__bf16)w0.w;
        Bs[(nc + 4) * 32 + kr] = (__bf16)w1.x;
        Bs[(nc + 5) * 32 + kr] = (__bf16)w1.y;
        Bs[(nc + 6) * 32 + kr] = (__bf16)w1.z;
        Bs[(nc + 7) * 32 + kr] = (__bf16)w1.w;
        __syncthreads();

        // ---- A fragment: two contiguous 8-float chunks, cvt to bf16 ----
        v16bf a;
        {
            const float4* a0 = (const float4*)(xr + kb + half * 8);
            const float4* a1 = (const float4*)(xr + kb + 16 + half * 8);
            cvt4(a, 0,  a0[0]);
            cvt4(a, 4,  a0[1]);
            cvt4(a, 8,  a1[0]);
            cvt4(a, 12, a1[1]);
        }

        // ---- 4 B fragments from LDS (ds_load_b128) + 4 WMMAs ----
        #pragma unroll
        for (int t = 0; t < 4; ++t) {
            const __bf16* brow = Bs + (t * 16 + nn) * 32;
            v8bf lo = *(const v8bf*)(brow + half * 8);
            v8bf hi = *(const v8bf*)(brow + 16 + half * 8);
            acc[t] = __builtin_amdgcn_wmma_f32_16x16x32_bf16(
                false, a, false, pack16(lo, hi), (short)0, acc[t], false, false);
        }
    }

    // ---- epilogue: bias + route to q/k (row-major) or v (transposed) ----
    const int which = ntile / HH;                    // 0=q 1=k 2=v (uniform)
    const int h     = ntile % HH;
    const int grow0 = mt * 128 + wid * 16 + half * 8;
    const int b  = grow0 / SS;
    const int s0 = grow0 % SS;

    #pragma unroll
    for (int t = 0; t < 4; ++t) {
        const int d    = t * 16 + nn;
        const float bias = bqkv[(size_t)ntile * 64 + t * 16 + nn];
        if (which == 2) {
            v8bf pk;
            #pragma unroll
            for (int r = 0; r < 8; ++r) pk[r] = (__bf16)(acc[t][r] + bias);
            *(v8bf*)(vT + ((size_t)(b * HH + h) * HD + d) * SS + s0) = pk;
        } else {
            __bf16* dst = (which == 0) ? qh : kh;
            #pragma unroll
            for (int r = 0; r < 8; ++r)
                dst[((size_t)(b * HH + h) * SS + s0 + r) * HD + d] =
                    (__bf16)(acc[t][r] + bias);
        }
    }
}

// ---------------------------------------------------------------------------
// Kernel 2: flash attention.  One wave per (b, h, 16-query tile).
// Block = 12 waves = all heads of one (b, qtile) -> identical trip counts.
// Each wave's 32x64 K block is DMA'd into a private LDS buffer by the
// Tensor Data Mover (tensor_load_to_lds + s_wait_tensorcnt); score and P.V
// fragments are aligned 128-bit LDS/global loads.
// ---------------------------------------------------------------------------
__global__ __launch_bounds__(384) void attn_kernel(
    const __bf16* __restrict__ qh, const __bf16* __restrict__ kh,
    const __bf16* __restrict__ vT, __bf16* __restrict__ ctx)
{
    const int lane = threadIdx.x & 31;
    const int h    = threadIdx.x >> 5;               // wave id == head
    const int b    = blockIdx.x >> 6;                // 64 query tiles
    const int qt   = blockIdx.x & 63;
    const int nn   = lane & 15;
    const int half = lane >> 4;

    __shared__ __align__(16) __bf16 pshare[HH][16 * 32];  // P relayout
    __shared__ __align__(16) __bf16 kshare[HH][32 * HD];  // TDM K staging
    __bf16* P  = pshare[h];
    __bf16* Ks = kshare[h];

    const size_t head_off = (size_t)(b * HH + h) * SS * HD;
    const __bf16* qb  = qh + head_off + (size_t)qt * 16 * HD;
    const __bf16* kb_ = kh + head_off;
    const __bf16* vb  = vT + head_off;               // [hd][S] transposed

    // Q fragments for d in [0,32) and [32,64): four b128 loads
    v16bf qa0, qa1;
    {
        const __bf16* qr = qb + (size_t)nn * HD;
        qa0 = pack16(*(const v8bf*)(qr + half * 8),
                     *(const v8bf*)(qr + 16 + half * 8));
        qa1 = pack16(*(const v8bf*)(qr + 32 + half * 8),
                     *(const v8bf*)(qr + 48 + half * 8));
    }

    float mrow[8], lrow[8];
    v8f o[4];
    {
        v8f z = {};
        #pragma unroll
        for (int t = 0; t < 4; ++t) o[t] = z;
        #pragma unroll
        for (int r = 0; r < 8; ++r) { mrow[r] = -3.0e38f; lrow[r] = 0.0f; }
    }

    const float scale = 0.125f;                      // 1/sqrt(64)
    const int nkb = (qt * 16 + 47) >> 5;             // causal #key-blocks of 32

    for (int jb = 0; jb < nkb; ++jb) {
        const int j0 = jb * 32;
        __builtin_prefetch(vb + (size_t)nn * SS + j0, 0, 0);

        // ---- TDM: DMA K block [32 keys x 64 dims] bf16 into this wave's LDS
#if HAVE_TDM
        asm volatile("s_wait_dscnt 0" ::: "memory");  // WAR vs prior ds reads
        tdm_load_2d_bf16(kb_ + (size_t)j0 * HD,
                         (unsigned int)(uintptr_t)Ks,
                         /*tensor_w=*/HD, /*tensor_h=*/SS,
                         /*tile_w=*/HD, /*tile_h=*/32, /*stride_e=*/HD);
        __builtin_amdgcn_s_wait_tensorcnt(0);
        const __bf16* kr0 = Ks + (size_t)nn * HD;
        const __bf16* kr1 = Ks + (size_t)(16 + nn) * HD;
#else
        const __bf16* kr0 = kb_ + (size_t)(j0 + nn) * HD;
        const __bf16* kr1 = kb_ + (size_t)(j0 + 16 + nn) * HD;
#endif

        // ---- scores: two 16x16 tiles, 4 WMMAs, b128 K fragment loads ----
        v8f s0 = {}, s1 = {};
        {
            v16bf bk0 = pack16(*(const v8bf*)(kr0 + half * 8),
                               *(const v8bf*)(kr0 + 16 + half * 8));
            v16bf bk1 = pack16(*(const v8bf*)(kr0 + 32 + half * 8),
                               *(const v8bf*)(kr0 + 48 + half * 8));
            s0 = __builtin_amdgcn_wmma_f32_16x16x32_bf16(false, qa0, false, bk0,
                                                         (short)0, s0, false, false);
            s0 = __builtin_amdgcn_wmma_f32_16x16x32_bf16(false, qa1, false, bk1,
                                                         (short)0, s0, false, false);
            v16bf bk2 = pack16(*(const v8bf*)(kr1 + half * 8),
                               *(const v8bf*)(kr1 + 16 + half * 8));
            v16bf bk3 = pack16(*(const v8bf*)(kr1 + 32 + half * 8),
                               *(const v8bf*)(kr1 + 48 + half * 8));
            s1 = __builtin_amdgcn_wmma_f32_16x16x32_bf16(false, qa0, false, bk2,
                                                         (short)0, s1, false, false);
            s1 = __builtin_amdgcn_wmma_f32_16x16x32_bf16(false, qa1, false, bk3,
                                                         (short)0, s1, false, false);
        }

        // ---- causal mask + online softmax (rows m = half*8 + r) ----
        float p0[8], p1[8];
        #pragma unroll
        for (int r = 0; r < 8; ++r) {
            const int q = qt * 16 + (half << 3) + r;
            float v0 = s0[r] * scale;
            float v1 = s1[r] * scale;
            if (j0 + nn > q)      v0 = -3.0e38f;
            if (j0 + 16 + nn > q) v1 = -3.0e38f;
            const float mx   = redmax16(fmaxf(v0, v1));
            const float mnew = fmaxf(mrow[r], mx);
            const float alpha = __expf(mrow[r] - mnew);
            p0[r] = __expf(v0 - mnew);
            p1[r] = __expf(v1 - mnew);
            const float rs = redsum16(p0[r] + p1[r]);
            lrow[r] = lrow[r] * alpha + rs;
            mrow[r] = mnew;
            #pragma unroll
            for (int t = 0; t < 4; ++t) o[t][r] *= alpha;
        }

        // ---- C-layout P -> LDS -> A-layout fragment (b128 LDS reads) ----
        #pragma unroll
        for (int r = 0; r < 8; ++r) {
            const int m = (half << 3) + r;
            P[m * 32 + nn]      = (__bf16)p0[r];
            P[m * 32 + 16 + nn] = (__bf16)p1[r];
        }
        const __bf16* prow = P + nn * 32;
        v16bf pa = pack16(*(const v8bf*)(prow + half * 8),
                          *(const v8bf*)(prow + 16 + half * 8));

        // ---- O += P (16x32) @ V (32x64): V^T rows are key-contiguous ----
        #pragma unroll
        for (int t = 0; t < 4; ++t) {
            const __bf16* vrow = vb + (size_t)(t * 16 + nn) * SS + j0;
            v16bf bv = pack16(*(const v8bf*)(vrow + half * 8),
                              *(const v8bf*)(vrow + 16 + half * 8));
            o[t] = __builtin_amdgcn_wmma_f32_16x16x32_bf16(false, pa, false, bv,
                                                           (short)0, o[t], false, false);
        }
    }

    // ---- normalize and store ctx as bf16 in merged-head [B,S,D] layout ----
    float inv[8];
    #pragma unroll
    for (int r = 0; r < 8; ++r) inv[r] = 1.0f / lrow[r];
    #pragma unroll
    for (int t = 0; t < 4; ++t) {
        #pragma unroll
        for (int r = 0; r < 8; ++r) {
            const size_t row = (size_t)b * SS + qt * 16 + (half << 3) + r;
            ctx[row * DD + h * HD + t * 16 + nn] = (__bf16)(o[t][r] * inv[r]);
        }
    }
}

// ---------------------------------------------------------------------------
// Kernel 3: out = ctx @ Wproj + bproj.  Same block-tiled structure as qkv:
// 128x64 tile per 8-wave block, LDS-transposed bf16 weight tile, b128 loads.
// ---------------------------------------------------------------------------
__global__ __launch_bounds__(256) void proj_gemm_kernel(
    const __bf16* __restrict__ ctx, const float* __restrict__ Wproj,
    const float* __restrict__ bproj, float* __restrict__ out)
{
    __shared__ __align__(16) __bf16 Bs[64 * 32];

    const int lane = threadIdx.x & 31;
    const int wid  = threadIdx.x >> 5;
    const int nn   = lane & 15;
    const int half = lane >> 4;
    const int ntile = blockIdx.x;                    // 0..11
    const int mt    = blockIdx.y;                    // 0..63

    const int kr = threadIdx.x >> 3;
    const int nc = (threadIdx.x & 7) << 3;
    const float* wsrc = Wproj + (size_t)ntile * 64 + nc;

    const int arowg = mt * 128 + wid * 16 + nn;
    const __bf16* cr = ctx + (size_t)arowg * DD;

    v8f acc[4];
    {
        v8f z = {};
        #pragma unroll
        for (int t = 0; t < 4; ++t) acc[t] = z;
    }

    for (int kb = 0; kb < DD; kb += 32) {
        const float* wrow = wsrc + (size_t)(kb + kr) * DD;
        float4 w0 = *(const float4*)(wrow);
        float4 w1 = *(const float4*)(wrow + 4);
        __syncthreads();
        Bs[(nc + 0) * 32 + kr] = (__bf16)w0.x;
        Bs[(nc + 1) * 32 + kr] = (__bf16)w0.y;
        Bs[(nc + 2) * 32 + kr] = (__bf16)w0.z;
        Bs[(nc + 3) * 32 + kr] = (__bf16)w0.w;
        Bs[(nc + 4) * 32 + kr] = (__bf16)w1.x;
        Bs[(nc + 5) * 32 + kr] = (__bf16)w1.y;
        Bs[(nc + 6) * 32 + kr] = (__bf16)w1.z;
        Bs[(nc + 7) * 32 + kr] = (__bf16)w1.w;
        __syncthreads();

        // A fragment: ctx already bf16, two b128 global loads
        v16bf a = pack16(*(const v8bf*)(cr + kb + half * 8),
                         *(const v8bf*)(cr + kb + 16 + half * 8));

        #pragma unroll
        for (int t = 0; t < 4; ++t) {
            const __bf16* brow = Bs + (t * 16 + nn) * 32;
            v8bf lo = *(const v8bf*)(brow + half * 8);
            v8bf hi = *(const v8bf*)(brow + 16 + half * 8);
            acc[t] = __builtin_amdgcn_wmma_f32_16x16x32_bf16(
                false, a, false, pack16(lo, hi), (short)0, acc[t], false, false);
        }
    }

    const int grow0 = mt * 128 + wid * 16 + half * 8;
    #pragma unroll
    for (int t = 0; t < 4; ++t) {
        const int ncol = ntile * 64 + t * 16 + nn;
        const float bias = bproj[ncol];
        #pragma unroll
        for (int r = 0; r < 8; ++r)
            out[(size_t)(grow0 + r) * DD + ncol] = acc[t][r] + bias;
    }
}

// ---------------------------------------------------------------------------
extern "C" void kernel_launch(void* const* d_in, const int* in_sizes, int n_in,
                              void* d_out, int out_size, void* d_ws, size_t ws_size,
                              hipStream_t stream) {
    const float* x     = (const float*)d_in[0];   // [8,1024,768]
    const float* Wqkv  = (const float*)d_in[1];   // [768,2304]
    const float* bqkv  = (const float*)d_in[2];   // [2304]
    const float* Wproj = (const float*)d_in[3];   // [768,768]
    const float* bproj = (const float*)d_in[4];   // [768]
    float* out = (float*)d_out;                   // [8,1024,768]

    const size_t nPer = (size_t)BB * HH * SS * HD;  // 6,291,456 (== B*S*D)
    char* ws = (char*)d_ws;
    __bf16* qh  = (__bf16*)(ws);
    __bf16* kh  = (__bf16*)(ws + 2 * nPer);
    __bf16* vT  = (__bf16*)(ws + 4 * nPer);        // transposed [B,H,hd,S]
    __bf16* ctx = (__bf16*)(ws + 6 * nPer);

    // 1) QKV GEMM: 36 col-tiles x 64 row-tiles, 256 threads (8 waves) each
    qkv_gemm_kernel<<<dim3(36, 64), 256, 0, stream>>>(x, Wqkv, bqkv, qh, kh, vT);

    // 2) Attention: 8 batches * 64 query tiles; 12 head-waves per block
    attn_kernel<<<BB * (SS / 16), 12 * 32, 0, stream>>>(qh, kh, vT, ctx);

    // 3) Projection GEMM: 12 col-tiles x 64 row-tiles
    proj_gemm_kernel<<<dim3(12, 64), 256, 0, stream>>>(ctx, Wproj, bproj, out);
}
